// CausalSelfAttention_27960237097604
// MI455X (gfx1250) — compile-verified, matrix operand order checked by reading, not run on hardware
//
#include <hip/hip_runtime.h>
#include <hip/hip_bf16.h>

// ---------------------------------------------------------------------------
// CDNA5 (gfx1250) causal self-attention, WMMA everywhere:
//   qkv = x @ w_attn ; flash-style causal attention ; out = y @ w_proj
// f16 operands, f32 accumulation via V_WMMA_F32_16X16X32_F16.
// All problem dims are compile-time powers of two so index math is shift/mask.
// ---------------------------------------------------------------------------

constexpr int BB = 2;      // batch
constexpr int TT = 2048;   // sequence length        (mask 2047, shift 11)
constexpr int CC = 1024;   // channels               (mask 1023, shift 10)
constexpr int HH = 16;     // heads                  (mask 15,   shift 4)
constexpr int DD = 64;     // head dim               (mask 63,   shift 6)
constexpr int MM = BB * TT;     // 4096 token rows
constexpr int N3 = 3 * CC;      // 3072

typedef __attribute__((ext_vector_type(16))) _Float16 v16h;  // WMMA A/B operand (8 VGPRs)
typedef __attribute__((ext_vector_type(8)))  _Float16 v8h;   // one b128 load
typedef __attribute__((ext_vector_type(8)))  float    v8f;   // WMMA C/D accumulator

static __device__ __forceinline__ v8f wmma_f16f32(v16h a, v16h b, v8f c) {
  // (neg_a, A, neg_b, B, c_mod, C, reuse_a, reuse_b)
  return __builtin_amdgcn_wmma_f32_16x16x32_f16(false, a, false, b, (short)0, c,
                                                false, false);
}

// Load a 16x32 (rows x K) f16 operand fragment from a row-major matrix.
// ISA A-layout (CDNA5 ISA): lanes 0-15 = rows, hold K={0..7,16..23};
// lanes 16-31 = same rows, hold K={8..15,24..31}.  Two b128 loads per lane.
// B operands use the same routine on a [N,K] (pre-transposed) matrix.
static __device__ __forceinline__ v16h frag_ld(const _Float16* base, int ld) {
  const int lane = threadIdx.x & 31;
  const int r    = lane & 15;
  const int ko   = (lane >> 4) << 3;  // 0 or 8
  const _Float16* p = base + (size_t)r * ld + ko;
  v8h lo = *(const v8h*)(p);        // K = ko .. ko+7
  v8h hi = *(const v8h*)(p + 16);   // K = ko+16 .. ko+23
  v16h f;
#pragma unroll
  for (int i = 0; i < 8; ++i) { f[i] = lo[i]; f[8 + i] = hi[i]; }
  return f;
}

// ---------------------------------------------------------------------------
// Precision conversion / weight transpose kernels
// ---------------------------------------------------------------------------
__global__ void k_f32_to_f16(const float* __restrict__ in,
                             _Float16* __restrict__ out, int n) {
  int i = blockIdx.x * blockDim.x + threadIdx.x;
  if (i < n) out[i] = (_Float16)in[i];
}

// in: [rows, cols] f32 row-major  ->  out: [cols, rows] f16 row-major
__global__ void k_transpose_f16(const float* __restrict__ in,
                                _Float16* __restrict__ out, int rows, int cols) {
  int i = blockIdx.x * blockDim.x + threadIdx.x;
  if (i < rows * cols) {
    int n = i / rows;   // output row (= input column)
    int k = i % rows;   // output col (= input row)
    out[(size_t)n * rows + k] = (_Float16)in[(size_t)k * cols + n];
  }
}

// ---------------------------------------------------------------------------
// WMMA GEMM #1:  qkv = xh[M,1024] @ w_attn  (BT = w_attn^T as [3072,1024])
// Block = 256 threads = 8 waves (2 x 4); block tile 64x256; wave tile 32x64.
// Per k-step: all 6 operand fragments (12 b128 loads) are materialized before
// the 8 WMMAs so the scheduler can stage s_wait_loadcnt instead of stalling.
// Epilogue scatters q,k -> [B,H,T,D] and v -> v^T [B,H,D,T] (shift/mask only).
// ---------------------------------------------------------------------------
__global__ void k_gemm_qkv(const _Float16* __restrict__ A,
                           const _Float16* __restrict__ BT,
                           _Float16* __restrict__ qh, _Float16* __restrict__ kh,
                           _Float16* __restrict__ vT) {
  constexpr int K = CC;
  const int wave = threadIdx.x >> 5;
  const int lane = threadIdx.x & 31;
  const int m0 = blockIdx.x * 64  + (wave >> 2) * 32;
  const int n0 = blockIdx.y * 256 + (wave & 3) * 64;

  v8f acc[2][4] = {};
  for (int k0 = 0; k0 < K; k0 += 32) {
    __builtin_prefetch(A  + (size_t)m0 * K + k0 + 64, 0, 1);
    __builtin_prefetch(BT + (size_t)n0 * K + k0 + 64, 0, 1);
    v16h a0 = frag_ld(A + (size_t)(m0     ) * K + k0, K);
    v16h a1 = frag_ld(A + (size_t)(m0 + 16) * K + k0, K);
    v16h b0 = frag_ld(BT + (size_t)(n0     ) * K + k0, K);
    v16h b1 = frag_ld(BT + (size_t)(n0 + 16) * K + k0, K);
    v16h b2 = frag_ld(BT + (size_t)(n0 + 32) * K + k0, K);
    v16h b3 = frag_ld(BT + (size_t)(n0 + 48) * K + k0, K);
    acc[0][0] = wmma_f16f32(a0, b0, acc[0][0]);
    acc[1][0] = wmma_f16f32(a1, b0, acc[1][0]);
    acc[0][1] = wmma_f16f32(a0, b1, acc[0][1]);
    acc[1][1] = wmma_f16f32(a1, b1, acc[1][1]);
    acc[0][2] = wmma_f16f32(a0, b2, acc[0][2]);
    acc[1][2] = wmma_f16f32(a1, b2, acc[1][2]);
    acc[0][3] = wmma_f16f32(a0, b3, acc[0][3]);
    acc[1][3] = wmma_f16f32(a1, b3, acc[1][3]);
  }

  // C/D layout: VGPR r -> (M = r + 8*(lane>>4), N = lane&15)
  const int mh = (lane >> 4) * 8;
  const int nc = lane & 15;
#pragma unroll
  for (int i = 0; i < 2; ++i)
#pragma unroll
    for (int j = 0; j < 4; ++j)
#pragma unroll
      for (int r = 0; r < 8; ++r) {
        int m = m0 + i * 16 + mh + r;   // token row in [0, B*T)
        int n = n0 + j * 16 + nc;       // column in [0, 3C)
        int b   = m >> 11;              // / TT
        int t   = m & (TT - 1);
        int sel = n >> 10;              // / CC : 0=q, 1=k, 2=v
        int c   = n & (CC - 1);
        int h   = c >> 6;               // / DD
        int d   = c & (DD - 1);
        _Float16 hv = (_Float16)acc[i][j][r];
        size_t bh = (size_t)((b << 4) + h);      // b*HH + h
        if (sel == 0)      qh[(bh * TT + t) * DD + d] = hv;
        else if (sel == 1) kh[(bh * TT + t) * DD + d] = hv;
        else               vT[(bh * DD + d) * TT + t] = hv;  // transposed
      }
}

// ---------------------------------------------------------------------------
// Flash-style causal attention.
// grid = (B*H, T/64), block = 128 (4 waves); each wave owns 16 query rows.
// Per 32-wide s-block: 4 WMMAs for S = q k^T, online softmax with width-16
// shuffle reductions, P relayout via per-wave LDS tile, 4 WMMAs for O += P v.
// ---------------------------------------------------------------------------
__global__ void k_attention(const _Float16* __restrict__ qh,
                            const _Float16* __restrict__ kh,
                            const _Float16* __restrict__ vT,
                            _Float16* __restrict__ yh) {
  __shared__ _Float16 pst[4][16 * 32];  // one 16x32 P tile per wave

  const int lane = threadIdx.x & 31;
  const int wave = threadIdx.x >> 5;
  const int bh = blockIdx.x;            // 0 .. B*H-1
  const int b  = bh >> 4;               // / HH
  const int h  = bh & (HH - 1);
  const int t0 = blockIdx.y * 64 + wave * 16;

  const _Float16* qbase = qh + (size_t)bh * TT * DD;
  const _Float16* kbase = kh + (size_t)bh * TT * DD;
  const _Float16* vbase = vT + (size_t)bh * DD * TT;

  // q fragments for K(d) = 0..31 and 32..63 -- live for whole s loop
  v16h qa0 = frag_ld(qbase + (size_t)t0 * DD +  0, DD);
  v16h qa1 = frag_ld(qbase + (size_t)t0 * DD + 32, DD);

  v8f o[4] = {};                        // O tile 16x64 (4 N-tiles of d)
  float mrow[8], lrow[8];
#pragma unroll
  for (int r = 0; r < 8; ++r) { mrow[r] = -1.0e30f; lrow[r] = 0.0f; }

  const int mh = (lane >> 4) * 8;
  const int nc = lane & 15;
  const float scale = 0.125f;           // 1/sqrt(64)
  _Float16* st = &pst[wave][0];

  for (int s0 = 0; s0 <= t0 + 15; s0 += 32) {
    // ---- S = q k^T : two 16x16 N-tiles over s, K accumulated over d=64 ----
    v8f sc0 = {}, sc1 = {};
    v16h kb0a = frag_ld(kbase + (size_t)(s0     ) * DD +  0, DD);
    v16h kb0b = frag_ld(kbase + (size_t)(s0     ) * DD + 32, DD);
    v16h kb1a = frag_ld(kbase + (size_t)(s0 + 16) * DD +  0, DD);
    v16h kb1b = frag_ld(kbase + (size_t)(s0 + 16) * DD + 32, DD);
    // prefetch next s-block's V tiles into cache while S is computed
    __builtin_prefetch(vbase + s0 + 32, 0, 1);
    sc0 = wmma_f16f32(qa0, kb0a, sc0);
    sc0 = wmma_f16f32(qa1, kb0b, sc0);
    sc1 = wmma_f16f32(qa0, kb1a, sc1);
    sc1 = wmma_f16f32(qa1, kb1b, sc1);

    // ---- scale + causal mask ----
#pragma unroll
    for (int r = 0; r < 8; ++r) {
      int trow = t0 + mh + r;
      float e0 = sc0[r] * scale;
      float e1 = sc1[r] * scale;
      if (s0 + nc      > trow) e0 = -1.0e30f;
      if (s0 + 16 + nc > trow) e1 = -1.0e30f;
      sc0[r] = e0; sc1[r] = e1;
    }

    // ---- online softmax ----
#pragma unroll
    for (int r = 0; r < 8; ++r) {
      float mx = fmaxf(sc0[r], sc1[r]);
#pragma unroll
      for (int off = 1; off < 16; off <<= 1)
        mx = fmaxf(mx, __shfl_xor(mx, off, 32));
      float mnew  = fmaxf(mrow[r], mx);
      float alpha = __expf(mrow[r] - mnew);
      float p0 = __expf(sc0[r] - mnew);
      float p1 = __expf(sc1[r] - mnew);
      float sum = p0 + p1;
#pragma unroll
      for (int off = 1; off < 16; off <<= 1)
        sum += __shfl_xor(sum, off, 32);
      lrow[r] = lrow[r] * alpha + sum;
      mrow[r] = mnew;
#pragma unroll
      for (int dt = 0; dt < 4; ++dt) o[dt][r] *= alpha;
      // stage P (f16) into LDS, converting C-layout -> row-major 16x32 tile
      st[(mh + r) * 32 + nc]      = (_Float16)p0;
      st[(mh + r) * 32 + 16 + nc] = (_Float16)p1;
    }
    asm volatile("s_wait_dscnt 0" ::: "memory");  // DS writes before DS reads

    // ---- O += P @ V  (v^T is [d, t] row-major -> K-contiguous B operand) ---
    // Materialize all four V fragments first so load waits can be staged.
    v16h pa  = frag_ld(st, 32);
    v16h vb0 = frag_ld(vbase + (size_t) 0 * TT + s0, TT);
    v16h vb1 = frag_ld(vbase + (size_t)16 * TT + s0, TT);
    v16h vb2 = frag_ld(vbase + (size_t)32 * TT + s0, TT);
    v16h vb3 = frag_ld(vbase + (size_t)48 * TT + s0, TT);
    o[0] = wmma_f16f32(pa, vb0, o[0]);
    o[1] = wmma_f16f32(pa, vb1, o[1]);
    o[2] = wmma_f16f32(pa, vb2, o[2]);
    o[3] = wmma_f16f32(pa, vb3, o[3]);
  }

  // ---- normalize + write y as f16 [B, T, C] for the projection GEMM ----
#pragma unroll
  for (int dt = 0; dt < 4; ++dt)
#pragma unroll
    for (int r = 0; r < 8; ++r) {
      int trow = t0 + mh + r;
      int col  = (h << 6) + dt * 16 + nc;           // h*DD + ...
      yh[((size_t)b * TT + trow) * CC + col] = (_Float16)(o[dt][r] / lrow[r]);
    }
}

// ---------------------------------------------------------------------------
// WMMA GEMM #2: out = yh[M,1024] @ w_proj  (BT = w_proj^T [1024,1024]).
// Same 64x256 block / 32x64 wave tiling; fp32 output straight to d_out.
// ---------------------------------------------------------------------------
__global__ void k_gemm_proj(const _Float16* __restrict__ A,
                            const _Float16* __restrict__ BT,
                            float* __restrict__ out) {
  constexpr int K = CC;
  const int wave = threadIdx.x >> 5;
  const int lane = threadIdx.x & 31;
  const int m0 = blockIdx.x * 64  + (wave >> 2) * 32;
  const int n0 = blockIdx.y * 256 + (wave & 3) * 64;

  v8f acc[2][4] = {};
  for (int k0 = 0; k0 < K; k0 += 32) {
    __builtin_prefetch(A  + (size_t)m0 * K + k0 + 64, 0, 1);
    __builtin_prefetch(BT + (size_t)n0 * K + k0 + 64, 0, 1);
    v16h a0 = frag_ld(A + (size_t)(m0     ) * K + k0, K);
    v16h a1 = frag_ld(A + (size_t)(m0 + 16) * K + k0, K);
    v16h b0 = frag_ld(BT + (size_t)(n0     ) * K + k0, K);
    v16h b1 = frag_ld(BT + (size_t)(n0 + 16) * K + k0, K);
    v16h b2 = frag_ld(BT + (size_t)(n0 + 32) * K + k0, K);
    v16h b3 = frag_ld(BT + (size_t)(n0 + 48) * K + k0, K);
    acc[0][0] = wmma_f16f32(a0, b0, acc[0][0]);
    acc[1][0] = wmma_f16f32(a1, b0, acc[1][0]);
    acc[0][1] = wmma_f16f32(a0, b1, acc[0][1]);
    acc[1][1] = wmma_f16f32(a1, b1, acc[1][1]);
    acc[0][2] = wmma_f16f32(a0, b2, acc[0][2]);
    acc[1][2] = wmma_f16f32(a1, b2, acc[1][2]);
    acc[0][3] = wmma_f16f32(a0, b3, acc[0][3]);
    acc[1][3] = wmma_f16f32(a1, b3, acc[1][3]);
  }

  const int mh = (lane >> 4) * 8;
  const int nc = lane & 15;
#pragma unroll
  for (int i = 0; i < 2; ++i)
#pragma unroll
    for (int j = 0; j < 4; ++j)
#pragma unroll
      for (int r = 0; r < 8; ++r) {
        int m = m0 + i * 16 + mh + r;
        int n = n0 + j * 16 + nc;
        out[(size_t)m * CC + n] = acc[i][j][r];
      }
}

// ---------------------------------------------------------------------------
// Host launcher
// ---------------------------------------------------------------------------
extern "C" void kernel_launch(void* const* d_in, const int* in_sizes, int n_in,
                              void* d_out, int out_size, void* d_ws, size_t ws_size,
                              hipStream_t stream) {
  (void)in_sizes; (void)n_in; (void)out_size; (void)ws_size;

  const float* x      = (const float*)d_in[0];
  const float* w_attn = (const float*)d_in[1];
  const float* w_proj = (const float*)d_in[2];
  float*       out    = (float*)d_out;

  // workspace carve-up (f16 buffers), 256B aligned
  char* ws = (char*)d_ws;
  size_t off = 0;
  auto take = [&](size_t bytes) -> char* {
    char* p = ws + off;
    off += (bytes + 255) & ~(size_t)255;
    return p;
  };
  _Float16* xh  = (_Float16*)take((size_t)MM * CC * 2);   // x in f16
  _Float16* waT = (_Float16*)take((size_t)N3 * CC * 2);   // w_attn^T [3C, C]
  _Float16* wpT = (_Float16*)take((size_t)CC * CC * 2);   // w_proj^T [C, C]
  _Float16* qh  = (_Float16*)take((size_t)BB * HH * TT * DD * 2);
  _Float16* kh  = (_Float16*)take((size_t)BB * HH * TT * DD * 2);
  _Float16* vT  = (_Float16*)take((size_t)BB * HH * DD * TT * 2);  // [B,H,D,T]
  _Float16* yh  = (_Float16*)take((size_t)MM * CC * 2);

  // 1) precision conversions / weight transposes
  {
    int n = MM * CC;
    k_f32_to_f16<<<(n + 255) / 256, 256, 0, stream>>>(x, xh, n);
  }
  {
    int n = CC * N3;
    k_transpose_f16<<<(n + 255) / 256, 256, 0, stream>>>(w_attn, waT, CC, N3);
  }
  {
    int n = CC * CC;
    k_transpose_f16<<<(n + 255) / 256, 256, 0, stream>>>(w_proj, wpT, CC, CC);
  }

  // 2) qkv projection (scatters q, k, v^T)
  k_gemm_qkv<<<dim3(MM / 64, N3 / 256), 256, 0, stream>>>(xh, waT, qh, kh, vT);

  // 3) causal flash attention
  k_attention<<<dim3(BB * HH, TT / 64), 128, 0, stream>>>(qh, kh, vT, yh);

  // 4) output projection -> fp32
  k_gemm_proj<<<dim3(MM / 64, CC / 256), 256, 0, stream>>>(yh, wpT, out);
}
